// PrefixDecoderBlock_50130858279070
// MI455X (gfx1250) — compile-verified, matrix operand order checked by reading, not run on hardware
//
#include <hip/hip_runtime.h>

typedef __bf16 bhalf_t;
typedef __attribute__((ext_vector_type(16))) __bf16 v16bf;
typedef __attribute__((ext_vector_type(8)))  __bf16 v8bf;
typedef __attribute__((ext_vector_type(8)))  float  v8f;
typedef __attribute__((ext_vector_type(4)))  unsigned int v4u;
typedef __attribute__((ext_vector_type(8)))  int v8i;
typedef __attribute__((ext_vector_type(4)))  int v4i;

#define DM   1024   // d_model
#define NH   16     // heads
#define HD   64     // head dim
#define DF   4096   // d_ff
#define BB   2      // batch
#define LL   2048   // seq len
#define MM   (BB*LL)

#if defined(__gfx1250__) && __has_builtin(__builtin_amdgcn_tensor_load_to_lds)
#define HAVE_TDM 1
#else
#define HAVE_TDM 0
#endif

// ---------------------------------------------------------------------------
// WMMA helpers
// ---------------------------------------------------------------------------
__device__ __forceinline__ v8f wmma_bf16f32(v16bf a, v16bf b, v8f c) {
  return __builtin_amdgcn_wmma_f32_16x16x32_bf16(false, a, false, b, (short)0, c,
                                                 false, false);
}

// Load one bf16 WMMA fragment (A or B, 16x32, row-major source).
// Caller passes p = rowBase + (lane>>4)*8; per the CDNA5 16-bit layout the
// 16 elements are K offsets {g*8..g*8+7, 16+g*8..23+g*8}: two 16B loads.
__device__ __forceinline__ v16bf frag_ld(const bhalf_t* p) {
  v8bf lo = *(const v8bf*)(p);
  v8bf hi = *(const v8bf*)(p + 16);
  v16bf r;
#pragma unroll
  for (int i = 0; i < 8; ++i) { r[i] = lo[i]; r[8 + i] = hi[i]; }
  return r;
}

#if HAVE_TDM
// ---------------------------------------------------------------------------
// Tensor Data Mover: DMA a 2D tile (tile_k contiguous bf16 elems x tile_rows
// rows, row pitch row_stride elems) from global into LDS (packed rows).
// D# packing per CDNA5 ISA ch.8 (group0 + group1; 2D tile -> groups 2/3 zero).
// ---------------------------------------------------------------------------
__device__ __forceinline__ void tdm_load_2d(unsigned lds_off, const void* gptr,
                                            unsigned tile_k, unsigned tile_rows,
                                            unsigned tensor_k, unsigned tensor_rows,
                                            unsigned long long row_stride) {
  unsigned long long ga = (unsigned long long)(uintptr_t)gptr;
  v4u g0;
  g0[0] = 1u;                                        // count=1, user descriptor
  g0[1] = lds_off;                                   // lds_addr (bytes)
  g0[2] = (unsigned)(ga & 0xFFFFFFFFu);              // global_addr[31:0]
  g0[3] = (unsigned)((ga >> 32) & 0x01FFFFFFu) | (2u << 30);  // addr[56:32], type=2
  v8i g1;
  g1[0] = (int)(1u << 16);                           // data_size=1 -> 2 bytes
  g1[1] = (int)((tensor_k & 0xFFFFu) << 16);         // tensor_dim0[15:0]
  g1[2] = (int)(((tensor_k >> 16) & 0xFFFFu) | ((tensor_rows & 0xFFFFu) << 16));
  g1[3] = (int)(((tensor_rows >> 16) & 0xFFFFu) | ((tile_k & 0xFFFFu) << 16));
  g1[4] = (int)(tile_rows & 0xFFFFu);                // tile_dim1 (tile_dim2=0)
  g1[5] = (int)(unsigned)(row_stride & 0xFFFFFFFFull);        // dim0_stride[31:0]
  g1[6] = (int)(unsigned)((row_stride >> 32) & 0xFFFFull);    // dim0_stride[47:32]
  g1[7] = 0;                                         // dim1_stride unused (2D tile)
  v4i z = {0, 0, 0, 0};
#if __clang_major__ >= 23
  v8i z8 = {0, 0, 0, 0, 0, 0, 0, 0};
  __builtin_amdgcn_tensor_load_to_lds(g0, g1, z, z, z8, 0);
#else
  __builtin_amdgcn_tensor_load_to_lds(g0, g1, z, z, 0);
#endif
}
#endif  // HAVE_TDM

// ---------------------------------------------------------------------------
// fp32 -> bf16 cast (weights)
// ---------------------------------------------------------------------------
__global__ void cast_f32_bf16(const float* __restrict__ in,
                              bhalf_t* __restrict__ out, int n) {
  int i = blockIdx.x * blockDim.x + threadIdx.x;
  if (i < n) out[i] = (bhalf_t)in[i];
}

// ---------------------------------------------------------------------------
// LayerNorm (d=1024) -> bf16
// ---------------------------------------------------------------------------
__global__ __launch_bounds__(256) void layernorm_bf16(
    const float* __restrict__ x, const float* __restrict__ w,
    const float* __restrict__ b, bhalf_t* __restrict__ out) {
  __shared__ float red[256];
  const int row = blockIdx.x;
  const int tid = threadIdx.x;
  const float* xr = x + (size_t)row * DM;
  float s = 0.f;
  for (int i = tid; i < DM; i += 256) s += xr[i];
  red[tid] = s; __syncthreads();
  for (int o = 128; o > 0; o >>= 1) {
    if (tid < o) red[tid] += red[tid + o];
    __syncthreads();
  }
  const float mu = red[0] * (1.f / DM);
  __syncthreads();
  float v = 0.f;
  for (int i = tid; i < DM; i += 256) { float d = xr[i] - mu; v += d * d; }
  red[tid] = v; __syncthreads();
  for (int o = 128; o > 0; o >>= 1) {
    if (tid < o) red[tid] += red[tid + o];
    __syncthreads();
  }
  const float rstd = rsqrtf(red[0] * (1.f / DM) + 1e-5f);
  bhalf_t* orow = out + (size_t)row * DM;
  for (int i = tid; i < DM; i += 256)
    orow[i] = (bhalf_t)((xr[i] - mu) * rstd * w[i] + b[i]);
}

// ---------------------------------------------------------------------------
// bf16 WMMA GEMM:  C[M,N] = epi( A[M,K] @ W[N,K]^T + bias[N] )
// 256 threads = 8 waves; wave tile 64x64 (4x4 WMMA tiles); block tile 128x256.
// TDM path: double-buffered LDS staging, tile DMA by TDM (TENSORcnt).
// EPI: 0 = bf16 store, 1 = bf16 V-transpose store [B,NH,HD,LL],
//      2 = fp32 residual-add store, 3 = bf16 exact-GELU store
// ---------------------------------------------------------------------------
template <int EPI>
__global__ __launch_bounds__(256) void gemm_bf16(
    const bhalf_t* __restrict__ A, const bhalf_t* __restrict__ W,
    const float* __restrict__ bias, const float* __restrict__ resid,
    void* __restrict__ outp, int Mdim, int Ndim, int Kdim) {
  const int lane = threadIdx.x & 31;
  const int wave = threadIdx.x >> 5;
  const int hl = lane & 15, g = lane >> 4;
  const int wm = wave & 1;            // 2 row groups of 64
  const int wn = wave >> 1;           // 4 col groups of 64
  const int m0 = blockIdx.x * 128 + wm * 64;
  const int n0 = blockIdx.y * 256 + wn * 64;

  v8f acc[4][4];
#pragma unroll
  for (int i = 0; i < 4; ++i)
#pragma unroll
    for (int j = 0; j < 4; ++j)
#pragma unroll
      for (int e = 0; e < 8; ++e) acc[i][j][e] = 0.f;

#if HAVE_TDM
  __shared__ bhalf_t sA[2][128][32];  // 16 KB: block A tile, double buffered
  __shared__ bhalf_t sB[2][256][32];  // 32 KB: block W tile, double buffered

  const bhalf_t* Ab = A + (size_t)blockIdx.x * 128 * Kdim;
  const bhalf_t* Wb = W + (size_t)blockIdx.y * 256 * Kdim;

  // prologue: DMA first K-slab into buffer 0 (one wave issues for the block)
  if (wave == 0) {
    tdm_load_2d((unsigned)(uintptr_t)&sA[0][0][0], Ab, 32, 128,
                (unsigned)Kdim, (unsigned)Mdim, (unsigned long long)Kdim);
    tdm_load_2d((unsigned)(uintptr_t)&sB[0][0][0], Wb, 32, 256,
                (unsigned)Kdim, (unsigned)Ndim, (unsigned long long)Kdim);
  }

  for (int k = 0; k < Kdim; k += 32) {
    const int buf = (k >> 5) & 1;
    if (wave == 0) __builtin_amdgcn_s_wait_tensorcnt(0);
    __syncthreads();  // buf ready for everyone; buf^1 free to overwrite
    if (k + 32 < Kdim && wave == 0) {
      tdm_load_2d((unsigned)(uintptr_t)&sA[buf ^ 1][0][0], Ab + (k + 32), 32, 128,
                  (unsigned)Kdim, (unsigned)Mdim, (unsigned long long)Kdim);
      tdm_load_2d((unsigned)(uintptr_t)&sB[buf ^ 1][0][0], Wb + (k + 32), 32, 256,
                  (unsigned)Kdim, (unsigned)Ndim, (unsigned long long)Kdim);
    }
    v16bf fa[4], fb[4];
#pragma unroll
    for (int mi = 0; mi < 4; ++mi)
      fa[mi] = frag_ld(&sA[buf][wm * 64 + mi * 16 + hl][g * 8]);
#pragma unroll
    for (int ni = 0; ni < 4; ++ni)
      fb[ni] = frag_ld(&sB[buf][wn * 64 + ni * 16 + hl][g * 8]);
#pragma unroll
    for (int mi = 0; mi < 4; ++mi)
#pragma unroll
      for (int ni = 0; ni < 4; ++ni)
        acc[mi][ni] = wmma_bf16f32(fa[mi], fb[ni], acc[mi][ni]);
  }
#else
  // fallback: fragments straight from global
  for (int k = 0; k < Kdim; k += 32) {
    v16bf fa[4], fb[4];
#pragma unroll
    for (int mi = 0; mi < 4; ++mi)
      fa[mi] = frag_ld(A + (size_t)(m0 + mi * 16 + hl) * Kdim + k + g * 8);
#pragma unroll
    for (int ni = 0; ni < 4; ++ni)
      fb[ni] = frag_ld(W + (size_t)(n0 + ni * 16 + hl) * Kdim + k + g * 8);
#pragma unroll
    for (int mi = 0; mi < 4; ++mi)
#pragma unroll
      for (int ni = 0; ni < 4; ++ni)
        acc[mi][ni] = wmma_bf16f32(fa[mi], fb[ni], acc[mi][ni]);
  }
#endif

#pragma unroll
  for (int mi = 0; mi < 4; ++mi)
#pragma unroll
    for (int ni = 0; ni < 4; ++ni) {
      const int col = n0 + ni * 16 + hl;
      const float bv = bias[col];
#pragma unroll
      for (int r = 0; r < 8; ++r) {
        const int row = m0 + mi * 16 + r + 8 * g;
        float v = acc[mi][ni][r] + bv;
        if (EPI == 0) {
          ((bhalf_t*)outp)[(size_t)row * Ndim + col] = (bhalf_t)v;
        } else if (EPI == 1) {
          const int bidx = row >> 11;  // row / LL
          const int seq = row & (LL - 1);
          const size_t idx =
              (((size_t)(bidx * NH + (col >> 6)) * HD) + (col & (HD - 1))) * LL + seq;
          ((bhalf_t*)outp)[idx] = (bhalf_t)v;
        } else if (EPI == 2) {
          const size_t idx = (size_t)row * Ndim + col;
          ((float*)outp)[idx] = resid[idx] + v;
        } else {
          const float gv = 0.5f * v * (1.f + erff(v * 0.70710678118654752f));
          ((bhalf_t*)outp)[(size_t)row * Ndim + col] = (bhalf_t)gv;
        }
      }
    }
}

// ---------------------------------------------------------------------------
// Flash-style prefix-decoder attention.
// One block = one (b,h) and 128 consecutive queries (8 waves x 16-query tile),
// so all 8 waves share each 32-key K/V chunk, staged by TDM into double-
// buffered LDS (8x cut in global K/V traffic vs per-wave streaming).
// Q,K: [B*L, DM] (head-contiguous).  Vt: [B*NH*HD, LL] (transposed V).
// ---------------------------------------------------------------------------
__global__ __launch_bounds__(256) void attn_flash(
    const bhalf_t* __restrict__ Q, const bhalf_t* __restrict__ Km,
    const bhalf_t* __restrict__ Vt, bhalf_t* __restrict__ ctxout,
    const int* __restrict__ pfx) {
#if HAVE_TDM
  __shared__ bhalf_t sK[2][32][64];   // 8 KB: 32 keys x 64 hd, double buffered
  __shared__ bhalf_t sV[2][64][32];   // 8 KB: 64 hd x 32 keys, double buffered
#endif
  __shared__ bhalf_t plds[8][16][32]; // per-wave prob tile (transpose buffer)
  const int prefix = *pfx;
  const int lane = threadIdx.x & 31;
  const int wave = threadIdx.x >> 5;
  const int hl = lane & 15, g = lane >> 4;
  const int qblk = (blockIdx.x & (LL / 128 - 1)) * 128;  // query block start
  const int bh = blockIdx.x >> 4;                        // LL/128 == 16
  const int bidx = bh >> 4, h = bh & (NH - 1);
  const int q0 = qblk + wave * 16;                       // this wave's tile

  const bhalf_t* Qb = Q + (size_t)bidx * LL * DM + h * HD;
  const bhalf_t* Kb = Km + (size_t)bidx * LL * DM + h * HD;
  const bhalf_t* Vb = Vt + (size_t)bh * HD * LL;

  const bhalf_t* qp = Qb + (size_t)(q0 + hl) * DM + g * 8;
  const v16bf qa0 = frag_ld(qp);
  const v16bf qa1 = frag_ld(qp + 32);

  float mrow[8], lrow[8];
  v8f ctx[4];
#pragma unroll
  for (int r = 0; r < 8; ++r) { mrow[r] = -1e30f; lrow[r] = 0.f; }
#pragma unroll
  for (int nt = 0; nt < 4; ++nt)
#pragma unroll
    for (int e = 0; e < 8; ++e) ctx[nt][e] = 0.f;

  // per-wave and block-level key extents (rounded to 32-key chunks)
  int kend_w = (q0 + 16 > prefix) ? (q0 + 16) : prefix;
  kend_w = (kend_w + 31) & ~31;
  int kend_b = (qblk + 128 > prefix) ? (qblk + 128) : prefix;
  kend_b = (kend_b + 31) & ~31;
  const float scale = 0.125f;  // 1/sqrt(64)

#if HAVE_TDM
  if (wave == 0) {  // prologue DMA: K chunk (32x64, pitch DM), V chunk (64x32, pitch LL)
    tdm_load_2d((unsigned)(uintptr_t)&sK[0][0][0], Kb, 64, 32, 64, 32,
                (unsigned long long)DM);
    tdm_load_2d((unsigned)(uintptr_t)&sV[0][0][0], Vb, 32, 64, 32, 64,
                (unsigned long long)LL);
  }
#endif

  for (int c = 0; c < kend_b; c += 32) {
#if HAVE_TDM
    const int buf = (c >> 5) & 1;
    if (wave == 0) __builtin_amdgcn_s_wait_tensorcnt(0);
    __syncthreads();  // chunk ready for all waves; other buffer free
    if (c + 32 < kend_b && wave == 0) {
      tdm_load_2d((unsigned)(uintptr_t)&sK[buf ^ 1][0][0], Kb + (size_t)(c + 32) * DM,
                  64, 32, 64, 32, (unsigned long long)DM);
      tdm_load_2d((unsigned)(uintptr_t)&sV[buf ^ 1][0][0], Vb + (c + 32),
                  32, 64, 32, 64, (unsigned long long)LL);
    }
#endif
    if (c < kend_w) {  // wave-uniform causal trip-count guard (EXEC stays full)
      // ---- scores S[16 x 32] = Q @ K^T (2 key sub-tiles x 2 hd halves)
      v8f s[2];
#pragma unroll
      for (int t = 0; t < 2; ++t) {
#pragma unroll
        for (int e = 0; e < 8; ++e) s[t][e] = 0.f;
#if HAVE_TDM
        v16bf kb0 = frag_ld(&sK[buf][t * 16 + hl][g * 8]);
        v16bf kb1 = frag_ld(&sK[buf][t * 16 + hl][32 + g * 8]);
#else
        const bhalf_t* kp = Kb + (size_t)(c + t * 16 + hl) * DM + g * 8;
        v16bf kb0 = frag_ld(kp);
        v16bf kb1 = frag_ld(kp + 32);
#endif
        s[t] = wmma_bf16f32(qa0, kb0, s[t]);
        s[t] = wmma_bf16f32(qa1, kb1, s[t]);
      }
      // ---- scale + prefix-decoder mask
      const bool needmask = (c + 31 >= prefix);
#pragma unroll
      for (int t = 0; t < 2; ++t)
#pragma unroll
        for (int r = 0; r < 8; ++r) {
          float v = s[t][r] * scale;
          if (needmask) {
            const int key = c + t * 16 + hl;
            const int qi = q0 + r + 8 * g;
            const bool ok = (key < prefix) || ((qi >= prefix) && (key <= qi));
            if (!ok) v = -1e30f;
          }
          s[t][r] = v;
        }
      // ---- online softmax (row stats live in each 16-lane half)
#pragma unroll
      for (int r = 0; r < 8; ++r) {
        float tm = fmaxf(s[0][r], s[1][r]);
#pragma unroll
        for (int o = 1; o < 16; o <<= 1) tm = fmaxf(tm, __shfl_xor(tm, o, 32));
        const float mnew = fmaxf(mrow[r], tm);
        const float alpha = __expf(mrow[r] - mnew);
        float rs = 0.f;
#pragma unroll
        for (int t = 0; t < 2; ++t) {
          const float p = __expf(s[t][r] - mnew);
          s[t][r] = p;
          rs += p;
        }
#pragma unroll
        for (int o = 1; o < 16; o <<= 1) rs += __shfl_xor(rs, o, 32);
        lrow[r] = lrow[r] * alpha + rs;
        mrow[r] = mnew;
#pragma unroll
        for (int nt = 0; nt < 4; ++nt) ctx[nt][r] *= alpha;
      }
      // ---- C-tile -> A-fragment relayout through per-wave LDS
#pragma unroll
      for (int t = 0; t < 2; ++t)
#pragma unroll
        for (int r = 0; r < 8; ++r)
          plds[wave][r + 8 * g][t * 16 + hl] = (bhalf_t)s[t][r];
      const v16bf pa = frag_ld(&plds[wave][hl][g * 8]);  // ds_load_b128 x2
      // ---- ctx += P @ V  (contraction over 32 keys, 4 hd column tiles)
#pragma unroll
      for (int nt = 0; nt < 4; ++nt) {
#if HAVE_TDM
        v16bf vb = frag_ld(&sV[buf][nt * 16 + hl][g * 8]);
#else
        v16bf vb = frag_ld(Vb + (size_t)(nt * 16 + hl) * LL + c + g * 8);
#endif
        ctx[nt] = wmma_bf16f32(pa, vb, ctx[nt]);
      }
    }
  }
  // ---- normalize and store context
#pragma unroll
  for (int nt = 0; nt < 4; ++nt)
#pragma unroll
    for (int r = 0; r < 8; ++r) {
      const int row = q0 + r + 8 * g;
      const int col = h * HD + nt * 16 + hl;
      ctxout[((size_t)bidx * LL + row) * DM + col] = (bhalf_t)(ctx[nt][r] / lrow[r]);
    }
}

// ---------------------------------------------------------------------------
// Launch
// ---------------------------------------------------------------------------
extern "C" void kernel_launch(void* const* d_in, const int* in_sizes, int n_in,
                              void* d_out, int out_size, void* d_ws, size_t ws_size,
                              hipStream_t stream) {
  (void)in_sizes; (void)n_in; (void)out_size; (void)ws_size;
  const float* x    = (const float*)d_in[0];
  const float* ln1w = (const float*)d_in[1];
  const float* ln1b = (const float*)d_in[2];
  const float* inw  = (const float*)d_in[3];
  const float* inb  = (const float*)d_in[4];
  const float* outw = (const float*)d_in[5];
  const float* outb = (const float*)d_in[6];
  const float* ln2w = (const float*)d_in[7];
  const float* ln2b = (const float*)d_in[8];
  const float* fc1w = (const float*)d_in[9];
  const float* fc1b = (const float*)d_in[10];
  const float* fc2w = (const float*)d_in[11];
  const float* fc2b = (const float*)d_in[12];
  const int*   pfx  = (const int*)d_in[13];
  float* out = (float*)d_out;

  char* ws = (char*)d_ws;
  size_t off = 0;
  auto carve = [&](size_t bytes) -> void* {
    void* p = ws + off;
    off += (bytes + 255) & ~(size_t)255;
    return p;
  };
  bhalf_t* wqkv = (bhalf_t*)carve((size_t)3 * DM * DM * sizeof(bhalf_t));
  bhalf_t* wo   = (bhalf_t*)carve((size_t)DM * DM * sizeof(bhalf_t));
  bhalf_t* w1   = (bhalf_t*)carve((size_t)DF * DM * sizeof(bhalf_t));
  bhalf_t* w2   = (bhalf_t*)carve((size_t)DM * DF * sizeof(bhalf_t));
  bhalf_t* h1   = (bhalf_t*)carve((size_t)MM * DM * sizeof(bhalf_t));
  bhalf_t* Qb   = (bhalf_t*)carve((size_t)MM * DM * sizeof(bhalf_t));
  bhalf_t* Kb   = (bhalf_t*)carve((size_t)MM * DM * sizeof(bhalf_t));
  bhalf_t* Vt   = (bhalf_t*)carve((size_t)MM * DM * sizeof(bhalf_t));
  bhalf_t* ctxb = (bhalf_t*)carve((size_t)MM * DM * sizeof(bhalf_t));
  float*   x1   = (float*)carve((size_t)MM * DM * sizeof(float));
  bhalf_t* h2   = (bhalf_t*)carve((size_t)MM * DM * sizeof(bhalf_t));
  bhalf_t* ff1  = (bhalf_t*)carve((size_t)MM * DF * sizeof(bhalf_t));

  auto castN = [&](const float* src, bhalf_t* dst, int n) {
    cast_f32_bf16<<<(n + 255) / 256, 256, 0, stream>>>(src, dst, n);
  };
  castN(inw,  wqkv, 3 * DM * DM);
  castN(outw, wo,   DM * DM);
  castN(fc1w, w1,   DF * DM);
  castN(fc2w, w2,   DM * DF);

  layernorm_bf16<<<MM, 256, 0, stream>>>(x, ln1w, ln1b, h1);

  dim3 blk(256);
  dim3 gD(MM / 128, DM / 256);
  dim3 gF(MM / 128, DF / 256);
  gemm_bf16<0><<<gD, blk, 0, stream>>>(h1, wqkv,                       inb,          nullptr, Qb, MM, DM, DM);
  gemm_bf16<0><<<gD, blk, 0, stream>>>(h1, wqkv + (size_t)DM * DM,     inb + DM,     nullptr, Kb, MM, DM, DM);
  gemm_bf16<1><<<gD, blk, 0, stream>>>(h1, wqkv + (size_t)2 * DM * DM, inb + 2 * DM, nullptr, Vt, MM, DM, DM);

  attn_flash<<<BB * NH * (LL / 128), blk, 0, stream>>>(Qb, Kb, Vt, ctxb, pfx);

  gemm_bf16<2><<<gD, blk, 0, stream>>>(ctxb, wo, outb, x, x1, MM, DM, DM);
  layernorm_bf16<<<MM, 256, 0, stream>>>(x1, ln2w, ln2b, h2);
  gemm_bf16<3><<<gF, blk, 0, stream>>>(h2, w1, fc1b, nullptr, ff1, MM, DF, DM);
  gemm_bf16<2><<<gD, blk, 0, stream>>>(ff1, w2, fc2b, x1, out, MM, DM, DF);
}
